// SpatialMultiAttention_77197742178875
// MI455X (gfx1250) — matrix operand order repaired and checked by reading, not compiled, not run on hardware
//
#include <hip/hip_runtime.h>
#include <hip/hip_bf16.h>
#include <hip/hip_fp16.h>

// ---------------------------------------------------------------------------
// CDNA5 (gfx1250) dual-branch spatial-reduction attention.
// All large matmuls via v_wmma_f32_16x16x32_f16 (f32 accumulation).
// B operands (weights, V) are pre-swizzled into WMMA fragment order so every
// B-fragment is one aligned 32B load (2x global_load_b128).
// ---------------------------------------------------------------------------

typedef __attribute__((ext_vector_type(16))) _Float16 v16h;
typedef __attribute__((ext_vector_type(8)))  float    v8f;

#define V8F_ZERO {0.f,0.f,0.f,0.f,0.f,0.f,0.f,0.f}
#define WMMA_F16(a, b, c) \
  __builtin_amdgcn_wmma_f32_16x16x32_f16(false, (a), false, (b), (short)0, (c), false, false)

// ---- WMMA fragment helpers (k = 16*(lane>>4) + s in A and B; not the ISA table order) ------
// A (16x32 f16, MxK): lane m=lane&15 holds row m; lane half selects K groups.
static __device__ __forceinline__ v16h load_a_frag(const _Float16* base, int lda) {
  const int lane = threadIdx.x & 31;
  const int m = lane & 15, hs = lane >> 4;
  v16h a;
#pragma unroll
  for (int j = 0; j < 8; ++j) {
    const int bk = 2 * j + 16 * hs;
    const _Float16* e = base + (size_t)m * lda + bk;
    a[2 * j]     = e[0];
    a[2 * j + 1] = e[1];
  }
  return a;
}

// Swizzled B storage: element (k, n) of a [K x 128] matrix lives at
//   ((k>>5)*8 + (n>>4))*512 + (((k>>4)&1)*16 + (n&15))*16 + (k&15)
// so one lane's 16 halves of a 32x16 tile are contiguous (32B-aligned).
static __device__ __forceinline__ size_t bswz_idx(int k, int n) {
  return ((size_t)(k >> 5) * 8 + (n >> 4)) * 512 +
         (size_t)((((k >> 4) & 1) * 16 + (n & 15)) * 16 + (k & 15));
}

static __device__ __forceinline__ v16h load_b_frag_swz(const _Float16* mat, int kt, int nt) {
  const int lane = threadIdx.x & 31;
  return *(const v16h*)(mat + (((size_t)kt * 8 + nt) * 32 + lane) * 16);
}

// ---------------------------------------------------------------------------
// Prep kernels
// ---------------------------------------------------------------------------
__global__ __launch_bounds__(256) void k_convert_f16(const float* __restrict__ src,
                                                     _Float16* __restrict__ dst, int n) {
  int t = blockIdx.x * 256 + threadIdx.x;
  if (t < n) dst[t] = (_Float16)src[t];
}

// w (O=128, I=128) f32 -> swizzled B (k=i, n=o), transpose fused in.
__global__ __launch_bounds__(256) void k_pack_w128(const float* __restrict__ w,
                                                   _Float16* __restrict__ wt) {
  int t = blockIdx.x * 256 + threadIdx.x;  // 16384
  int o = t >> 7, i = t & 127;
  wt[bswz_idx(i, o)] = (_Float16)w[o * 128 + i];
}

// sr_w (O=128, I=128, ksr, ksr) f32 -> swizzled B with k=(kh*ksr+kw)*128+i, n=o.
__global__ __launch_bounds__(256) void k_pack_srw(const float* __restrict__ w,
                                                  _Float16* __restrict__ wt,
                                                  int kk, int total) {
  int t = blockIdx.x * 256 + threadIdx.x;
  if (t >= total) return;
  int o = t / (128 * kk);
  int rem = t - o * (128 * kk);
  int i = rem / kk;
  int p = rem - i * kk;  // kh*ksr + kw
  wt[bswz_idx(p * 128 + i, o)] = (_Float16)w[t];
}

// ---------------------------------------------------------------------------
// Generic GEMM: [M x 128] (f16) @ swizzled [128 x 128] -> f32 or f16 (+bias)
// Epilogue configuration is compile-time to keep the hot loop branch-free.
// 8 waves per block, one 16x16 tile per wave; blockIdx.x = row block.
// ---------------------------------------------------------------------------
template <bool OUTF, bool BIAS>
__global__ __launch_bounds__(256) void k_gemm128(const _Float16* __restrict__ A,
                                                 const _Float16* __restrict__ Bw,
                                                 float* __restrict__ outF,
                                                 _Float16* __restrict__ outH,
                                                 const float* __restrict__ bias) {
  const int wave = threadIdx.x >> 5;
  const int row0 = blockIdx.x * 16;
  const _Float16* a0 = A + (size_t)row0 * 128;
  v8f c = V8F_ZERO;
#pragma unroll
  for (int k0 = 0; k0 < 128; k0 += 32) {
    __builtin_prefetch(a0 + k0 + 512, 0, 3);  // global_prefetch_b8 (near)
    v16h a = load_a_frag(a0 + k0, 128);
    v16h b = load_b_frag_swz(Bw, k0 >> 5, wave);
    c = WMMA_F16(a, b, c);
  }
  const int lane = threadIdx.x & 31;
  const int col = (lane & 15) + wave * 16;
  const int hs = lane >> 4;
  float badd = 0.f;
  if constexpr (BIAS) badd = bias[col];
#pragma unroll
  for (int r = 0; r < 8; ++r) {
    const int row = row0 + r + 8 * hs;
    const float v = c[r] + badd;
    if constexpr (OUTF) outF[(size_t)row * 128 + col] = v;
    else                outH[(size_t)row * 128 + col] = (_Float16)v;
  }
}

// ---------------------------------------------------------------------------
// SR conv (kernel==stride) as implicit-im2col GEMM, fused LayerNorm + GELU.
// Each block: 16 output pixels x all 128 output channels (8 waves).
// A K-chunk of 32 never crosses a (kh,kw) patch position (128 % 32 == 0), so
// the patch decode is hoisted per chunk and fragment loads stay contiguous.
// ---------------------------------------------------------------------------
__global__ __launch_bounds__(256) void k_conv_ln_gelu(const _Float16* __restrict__ xh,
                                                      const _Float16* __restrict__ Wt,
                                                      const float* __restrict__ bias,
                                                      const float* __restrict__ gamma,
                                                      const float* __restrict__ beta,
                                                      _Float16* __restrict__ out,
                                                      int ksr, int ls, int wk,
                                                      int npix, int Kred) {
  __shared__ float tile[16][128];
  __shared__ float mu[16], ri[16];
  const int wave = threadIdx.x >> 5;
  const int lane = threadIdx.x & 31;
  const int m = lane & 15, hs = lane >> 4;
  const int p = blockIdx.x * 16 + m;
  const int b = p / npix;
  const int rem = p - b * npix;
  const int oh = rem / wk;
  const int ow = rem - oh * wk;
  // x as (B, H*W, C) in f16; W == C == 128.
  const _Float16* rowbase =
      xh + ((size_t)b * 16384 + (size_t)(oh * ksr) * 128 + (size_t)(ow * ksr)) * 128;
  v8f c = V8F_ZERO;
  for (int k0 = 0; k0 < Kred; k0 += 32) {
    const int kh = k0 >> (7 + ls);
    const int kw = (k0 >> 7) & (ksr - 1);
    const int i0 = k0 & 127;
    const _Float16* cb = rowbase + (size_t)kh * 16384 + (size_t)kw * 128 + i0;
    __builtin_prefetch(cb + 512, 0, 3);
    v16h a;
#pragma unroll
    for (int j = 0; j < 8; ++j) {
      const int bk = 2 * j + 16 * hs;
      a[2 * j]     = cb[bk];
      a[2 * j + 1] = cb[bk + 1];
    }
    v16h bf = load_b_frag_swz(Wt, k0 >> 5, wave);
    c = WMMA_F16(a, bf, c);
  }
  const int col = (lane & 15) + wave * 16;
#pragma unroll
  for (int r = 0; r < 8; ++r) tile[r + 8 * hs][col] = c[r] + bias[col];
  __syncthreads();
  {
    // LayerNorm stats: 16 threads per row, shfl tree within 16-lane halves.
    const int row = threadIdx.x >> 4, sub = threadIdx.x & 15;
    float s = 0.f;
#pragma unroll
    for (int j = 0; j < 8; ++j) s += tile[row][sub * 8 + j];
#pragma unroll
    for (int o = 1; o < 16; o <<= 1) s += __shfl_xor(s, o, 32);
    const float mean = s * (1.f / 128.f);
    float v2 = 0.f;
#pragma unroll
    for (int j = 0; j < 8; ++j) {
      const float d = tile[row][sub * 8 + j] - mean;
      v2 += d * d;
    }
#pragma unroll
    for (int o = 1; o < 16; o <<= 1) v2 += __shfl_xor(v2, o, 32);
    if (sub == 0) {
      mu[row] = mean;
      ri[row] = rsqrtf(v2 * (1.f / 128.f) + 1e-5f);
    }
  }
  __syncthreads();
  for (int e = threadIdx.x; e < 2048; e += 256) {
    const int rr = e >> 7, cc = e & 127;
    float v = (tile[rr][cc] - mu[rr]) * ri[rr] * gamma[cc] + beta[cc];
    float g = 0.5f * v * (1.f + erff(v * 0.70710678118654752f));
    out[((size_t)blockIdx.x * 16 + rr) * 128 + cc] = (_Float16)g;
  }
}

// ---------------------------------------------------------------------------
// Depthwise 3x3 local conv on V (+ identity); emit K (f16, stride 64) and V
// pre-swizzled per head in WMMA B-fragment order:
//   vsw[((b*4 + h)*(Nk/32) + key>>5)*512 + (((key>>4)&1)*16 + d)*16 + (key&15)]
// ---------------------------------------------------------------------------
__global__ __launch_bounds__(256) void k_localconv(const float* __restrict__ kv,
                                                   const float* __restrict__ lw,
                                                   const float* __restrict__ lb,
                                                   _Float16* __restrict__ kh,
                                                   _Float16* __restrict__ vsw,
                                                   int hk, int wk) {
  const int t = blockIdx.x * 256 + threadIdx.x;
  const int c = t & 63;
  const int p = t >> 6;
  const int npix = hk * wk;
  const int b = p / npix;
  const int key = p - b * npix;
  const int y = key / wk;
  const int x = key - y * wk;
  kh[(size_t)p * 64 + c] = (_Float16)kv[(size_t)p * 128 + c];
  float acc = lb[c];
#pragma unroll
  for (int dy = 0; dy < 3; ++dy) {
#pragma unroll
    for (int dx = 0; dx < 3; ++dx) {
      const int yy = y + dy - 1, xx = x + dx - 1;
      if (yy >= 0 && yy < hk && xx >= 0 && xx < wk)
        acc += lw[c * 9 + dy * 3 + dx] *
               kv[((size_t)(b * npix + yy * wk + xx)) * 128 + 64 + c];
    }
  }
  const float vout = kv[(size_t)p * 128 + 64 + c] + acc;
  const int h = c >> 4, d = c & 15;
  const size_t vidx = ((size_t)(b * 4 + h) * (npix >> 5) + (key >> 5)) * 512 +
                      (size_t)((((key >> 4) & 1) * 16 + d) * 16 + (key & 15));
  vsw[vidx] = (_Float16)vout;
}

// ---------------------------------------------------------------------------
// Flash attention: one wave per (batch, head, 16-query block).
// QK^T: hd=16 zero-padded to K=32; P.V: full K=32 chunks over keys.
// ---------------------------------------------------------------------------
__global__ __launch_bounds__(128) void k_attn(const _Float16* __restrict__ qh,
                                              const _Float16* __restrict__ kh,
                                              const _Float16* __restrict__ vsw,
                                              _Float16* __restrict__ outh,
                                              int Nk, int head_off, int out_off) {
  __shared__ _Float16 Pst[4][16 * 32];
  const int wave = threadIdx.x >> 5;
  const int lane = threadIdx.x & 31;
  const int m = lane & 15, hs = lane >> 4;
  const int wid = blockIdx.x * 4 + wave;
  const int b = wid >> 12;          // 4096 waves per batch (4 heads * 1024 qblocks)
  const int hb = (wid >> 10) & 3;   // head within branch
  const int qblk = wid & 1023;
  const size_t qrow0 = (size_t)b * 16384 + (size_t)qblk * 16;
  const int qcol = (head_off + hb) * 16;
  const size_t krow0 = (size_t)b * Nk;
  const int kcol = hb * 16;
  const _Float16* vbase = vsw + (size_t)(b * 4 + hb) * (Nk >> 5) * 512;

  // Q fragment, K padded 16 -> 32 with zeros (lanes 16-31 are K>=16).
  v16h qa;
#pragma unroll
  for (int j = 0; j < 8; ++j) {
    const int bk = 2 * j + 16 * hs;
    if (bk < 16) {
      const _Float16* e = qh + (qrow0 + m) * 128 + qcol + bk;
      qa[2 * j]     = e[0];
      qa[2 * j + 1] = e[1];
    } else {
      qa[2 * j]     = (_Float16)0.f;
      qa[2 * j + 1] = (_Float16)0.f;
    }
  }

  float mrun[8], lrun[8];
#pragma unroll
  for (int r = 0; r < 8; ++r) { mrun[r] = -3.0e38f; lrun[r] = 0.f; }
  v8f acc  = V8F_ZERO;
  const v8f zero = V8F_ZERO;
  _Float16* pst = &Pst[wave][0];

  for (int kc = 0; kc < Nk; kc += 32) {
    // B frags for S = q.k^T: element (k=dim, n=key) = K[key][dim]; per-lane
    // contiguous in dim (merges to b128). Lanes 16-31 (K>=16) are padding.
    v16h kb0, kb1;
    const _Float16* kb = kh + (krow0 + kc + (lane & 15)) * 64 + kcol;
#pragma unroll
    for (int j = 0; j < 8; ++j) {
      kb0[2 * j]     = hs ? (_Float16)0.f : kb[2 * j];
      kb0[2 * j + 1] = hs ? (_Float16)0.f : kb[2 * j + 1];
      kb1[2 * j]     = hs ? (_Float16)0.f : kb[16 * 64 + 2 * j];
      kb1[2 * j + 1] = hs ? (_Float16)0.f : kb[16 * 64 + 2 * j + 1];
    }
    v8f s0 = WMMA_F16(qa, kb0, zero);
    v8f s1 = WMMA_F16(qa, kb1, zero);

#pragma unroll
    for (int r = 0; r < 8; ++r) {
      float x0 = s0[r] * 0.25f;   // scale = hd^-0.5
      float x1 = s1[r] * 0.25f;
      float rx = fmaxf(x0, x1);
#pragma unroll
      for (int o = 1; o < 16; o <<= 1) rx = fmaxf(rx, __shfl_xor(rx, o, 32));
      const float mn = fmaxf(mrun[r], rx);
      const float al = __expf(mrun[r] - mn);
      const float p0 = __expf(x0 - mn);
      const float p1 = __expf(x1 - mn);
      float rs = p0 + p1;
#pragma unroll
      for (int o = 1; o < 16; o <<= 1) rs += __shfl_xor(rs, o, 32);
      lrun[r] = lrun[r] * al + rs;
      mrun[r] = mn;
      acc[r] *= al;
      const int prow = r + 8 * hs;
      pst[prow * 32 + (lane & 15)]      = (_Float16)p0;
      pst[prow * 32 + 16 + (lane & 15)] = (_Float16)p1;
    }
    asm volatile("s_wait_dscnt 0" ::: "memory");
    v16h pa = load_a_frag(pst, 32);
    v16h vb = *(const v16h*)(vbase + ((size_t)(kc >> 5) * 32 + lane) * 16);
    acc = WMMA_F16(pa, vb, acc);
  }

#pragma unroll
  for (int r = 0; r < 8; ++r) {
    const float ov = acc[r] / lrun[r];
    outh[(qrow0 + r + 8 * hs) * 128 + out_off + hb * 16 + (lane & 15)] = (_Float16)ov;
  }
}

// ---------------------------------------------------------------------------
// Host launcher
// ---------------------------------------------------------------------------
extern "C" void kernel_launch(void* const* d_in, const int* in_sizes, int n_in,
                              void* d_out, int out_size, void* d_ws, size_t ws_size,
                              hipStream_t stream) {
  const float* x      = (const float*)d_in[0];
  const float* q_w    = (const float*)d_in[1];
  const float* sr1_w  = (const float*)d_in[2];
  const float* sr1_b  = (const float*)d_in[3];
  const float* n1g    = (const float*)d_in[4];
  const float* n1b    = (const float*)d_in[5];
  const float* sr2_w  = (const float*)d_in[6];
  const float* sr2_b  = (const float*)d_in[7];
  const float* n2g    = (const float*)d_in[8];
  const float* n2b    = (const float*)d_in[9];
  const float* kv1_w  = (const float*)d_in[10];
  const float* kv2_w  = (const float*)d_in[11];
  const float* lc1_w  = (const float*)d_in[12];
  const float* lc1_b  = (const float*)d_in[13];
  const float* lc2_w  = (const float*)d_in[14];
  const float* lc2_b  = (const float*)d_in[15];
  const float* proj_w = (const float*)d_in[16];
  const float* proj_b = (const float*)d_in[17];
  float* out = (float*)d_out;

  // Workspace carve-up (all 256B-aligned chunks).
  char* ws = (char*)d_ws;
  size_t off = 0;
  auto alloc = [&](size_t bytes) { char* p = ws + off; off += (bytes + 255) & ~(size_t)255; return p; };
  _Float16* xh     = (_Float16*)alloc(4194304 * 2);  // x f16       (B*N,128)
  _Float16* qh     = (_Float16*)alloc(4194304 * 2);  // q f16       (B*N,128)
  _Float16* ao     = (_Float16*)alloc(4194304 * 2);  // attn out    (B*N,128)
  _Float16* qwS    = (_Float16*)alloc(16384 * 2);    // swizzled B mats
  _Float16* kv1S   = (_Float16*)alloc(16384 * 2);
  _Float16* kv2S   = (_Float16*)alloc(16384 * 2);
  _Float16* projS  = (_Float16*)alloc(16384 * 2);
  _Float16* sr1S   = (_Float16*)alloc(1048576 * 2);  // [8192,128] swizzled
  _Float16* sr2S   = (_Float16*)alloc(262144 * 2);   // [2048,128] swizzled
  _Float16* xr1h   = (_Float16*)alloc(65536 * 2);    // (512,128)
  _Float16* xr2h   = (_Float16*)alloc(262144 * 2);   // (2048,128)
  float*    kv1f   = (float*)alloc(65536 * 4);
  float*    kv2f   = (float*)alloc(262144 * 4);
  _Float16* kh1    = (_Float16*)alloc(32768 * 2);    // K only, stride 64
  _Float16* kh2    = (_Float16*)alloc(131072 * 2);
  _Float16* vsw1   = (_Float16*)alloc(32768 * 2);    // V swizzled per head
  _Float16* vsw2   = (_Float16*)alloc(131072 * 2);

  // 1) prep: f16 conversion + swizzled weight packs
  k_convert_f16<<<16384, 256, 0, stream>>>(x, xh, 4194304);
  k_pack_w128<<<64, 256, 0, stream>>>(q_w, qwS);
  k_pack_w128<<<64, 256, 0, stream>>>(kv1_w, kv1S);
  k_pack_w128<<<64, 256, 0, stream>>>(kv2_w, kv2S);
  k_pack_w128<<<64, 256, 0, stream>>>(proj_w, projS);
  k_pack_srw<<<4096, 256, 0, stream>>>(sr1_w, sr1S, 64, 1048576);
  k_pack_srw<<<1024, 256, 0, stream>>>(sr2_w, sr2S, 16, 262144);

  // 2) q projection: (32768,128) @ (128,128), f16 out, no bias
  k_gemm128<false, false><<<2048, 256, 0, stream>>>(xh, qwS, nullptr, qh, nullptr);

  // 3) SR convs + LN + GELU  (branch1: 8x8/8 -> 16x16; branch2: 4x4/4 -> 32x32)
  k_conv_ln_gelu<<<32, 256, 0, stream>>>(xh, sr1S, sr1_b, n1g, n1b, xr1h, 8, 3, 16, 256, 8192);
  k_conv_ln_gelu<<<128, 256, 0, stream>>>(xh, sr2S, sr2_b, n2g, n2b, xr2h, 4, 2, 32, 1024, 2048);

  // 4) kv projections (f32 out, needed by local conv; no bias)
  k_gemm128<true, false><<<32, 256, 0, stream>>>(xr1h, kv1S, kv1f, nullptr, nullptr);
  k_gemm128<true, false><<<128, 256, 0, stream>>>(xr2h, kv2S, kv2f, nullptr, nullptr);

  // 5) depthwise 3x3 local conv on V; pack K (stride 64) + swizzled V
  k_localconv<<<128, 256, 0, stream>>>(kv1f, lc1_w, lc1_b, kh1, vsw1, 16, 16);
  k_localconv<<<512, 256, 0, stream>>>(kv2f, lc2_w, lc2_b, kh2, vsw2, 32, 32);

  // 6) flash attention per branch (branch1: heads 0-3 -> cols 0-63; branch2: 4-7 -> 64-127)
  k_attn<<<2048, 128, 0, stream>>>(qh, kh1, vsw1, ao, 256, 0, 0);
  k_attn<<<2048, 128, 0, stream>>>(qh, kh2, vsw2, ao, 1024, 4, 64);

  // 7) output projection with bias -> f32 d_out
  k_gemm128<true, true><<<2048, 256, 0, stream>>>(ao, projS, out, nullptr, proj_b);
}
